// BLSTM_62586263437991
// MI455X (gfx1250) — matrix operand order repaired and checked by reading, not compiled, not run on hardware
//
#include <hip/hip_runtime.h>
#include <hip/hip_bf16.h>

// ---------------------------------------------------------------------------
// Bidirectional LSTM for MI455X (gfx1250), wave32 + WMMA bf16.
//
// Phase 1: pack weights/x to bf16 in WMMA fragment order.
// Phase 2: xg = x @ kernel  (big parallel WMMA GEMM, f32 accum, stored in
//          C-fragment order so phase 3 loads it directly as the accumulator).
// Phase 3: persistent recurrent kernels: 4 workgroups total
//          (2 directions x 2 batch-halves; batches are recurrence-independent).
//          h kept in LDS in A-fragment layout; c in registers; rk in L2;
//          hardware v_tanh_f32 / v_med3_f32 for the gate math.
// ---------------------------------------------------------------------------

typedef __attribute__((ext_vector_type(16))) __bf16 v16bf;
typedef __attribute__((ext_vector_type(8)))  float  v8f;

#define B_   32
#define T_   1024
#define D_   256
#define U_   256
#define G4   1024   // 4*U columns per direction
#define NT   64     // N tiles per direction (1024/16)
#define KT   8      // K tiles (256/32)
#define MT   2      // M tiles (32/16)
#define GPITCH 20   // gS pitch: 16 local batches + 4 pad floats (80B, 16B-aligned)

// A-matrix (16-bit, 16x32) per-lane K offset for element e (0..15):
__device__ __forceinline__ int a_koff(int lane, int e) {
    return e + ((e >= 8) ? 8 : 0) + ((lane >= 16) ? 8 : 0);
}
// B-matrix (16-bit, 32x16) per-lane K offset:
__device__ __forceinline__ int b_koff(int lane, int e) {
    return e + ((e >= 8) ? 8 : 0) + ((lane >= 16) ? 8 : 0);
}

__device__ __forceinline__ float hard_sig(float v) {
    // clamp(0.2v+0.5, 0, 1) -> v_fma + v_med3
    return __builtin_amdgcn_fmed3f(__builtin_fmaf(0.2f, v, 0.5f), 0.f, 1.f);
}

__device__ __forceinline__ float fast_tanh(float v) {
#if __has_builtin(__builtin_amdgcn_tanhf)
    return __builtin_amdgcn_tanhf(v);          // v_tanh_f32 (gfx1250 TRANS op)
#elif __has_builtin(__builtin_amdgcn_tanh_f32)
    return __builtin_amdgcn_tanh_f32(v);
#else
    return tanhf(v);
#endif
}

// --- pack kernel / recurrent_kernel (f32 [256,2048]) into bf16 B-fragments --
// layout: [dir][nt][kt][lane][e]  (per matrix)
__global__ void pack_w_kernel(const float* __restrict__ k,
                              const float* __restrict__ rk,
                              __bf16* __restrict__ kpack,
                              __bf16* __restrict__ rkpack) {
    int gid   = blockIdx.x * blockDim.x + threadIdx.x;   // 0 .. 65535
    int which = gid >> 15;                               // 0=kernel, 1=recurrent
    int slot  = gid & 32767;                             // ((dir*64+nt)*8+kt)*32+lane
    int lane  = slot & 31;
    int kt    = (slot >> 5) & 7;
    int nt    = (slot >> 8) & 63;
    int dir   = (slot >> 14) & 1;
    const float* W = which ? rk : k;
    __bf16*      P = which ? rkpack : kpack;
    int col  = dir * G4 + nt * 16 + (lane & 15);
    int base = (((dir * NT + nt) * KT + kt) * 32 + lane) * 16;
#pragma unroll
    for (int e = 0; e < 16; ++e) {
        int kk = kt * 32 + b_koff(lane, e);
        P[base + e] = (__bf16)W[kk * (8 * U_) + col];
    }
}

// --- pack x (f32 [B,T,D]) into bf16 A-fragments: [t][mt][kt][lane][e] -------
__global__ void pack_x_kernel(const float* __restrict__ x,
                              __bf16* __restrict__ xA) {
    int gid  = blockIdx.x * blockDim.x + threadIdx.x;    // T*2*8*32 = 524288
    int lane = gid & 31;
    int kt   = (gid >> 5) & 7;
    int mt   = (gid >> 8) & 1;
    int t    = (gid >> 9);
    int b    = mt * 16 + (lane & 15);
    int base = (((t * MT + mt) * KT + kt) * 32 + lane) * 16;
    const float* xr = x + ((size_t)b * T_ + t) * D_ + kt * 32;
#pragma unroll
    for (int e = 0; e < 16; ++e)
        xA[base + e] = (__bf16)xr[a_koff(lane, e)];
}

// --- xg = x @ kernel, both directions; output f32 in C-fragment order -------
// xg layout: [dir][t][mt][nt][lane][r(8)]
__global__ __launch_bounds__(256)
void xg_gemm_kernel(const __bf16* __restrict__ xA,
                    const __bf16* __restrict__ kpack,
                    float* __restrict__ xg) {
    const int t    = blockIdx.x;
    const int dir  = blockIdx.y;
    const int lane = threadIdx.x & 31;
    const int w    = threadIdx.x >> 5;   // 8 waves; wave handles nt = w*8..w*8+7
    for (int mt = 0; mt < MT; ++mt) {
        for (int j = 0; j < 8; ++j) {
            int nt = w * 8 + j;
            v8f acc = {};
#pragma unroll
            for (int kt = 0; kt < KT; ++kt) {
                v16bf a = *(const v16bf*)(xA +
                    (((size_t)(t * MT + mt) * KT + kt) * 32 + lane) * 16);
                v16bf b = *(const v16bf*)(kpack +
                    (((size_t)(dir * NT + nt) * KT + kt) * 32 + lane) * 16);
                acc = __builtin_amdgcn_wmma_f32_16x16x32_bf16(
                        false, a, false, b, (short)0, acc, false, false);
            }
            *(v8f*)(xg + (((((size_t)dir * T_ + t) * MT + mt) * NT + nt) * 32 + lane) * 8) = acc;
        }
    }
}

// --- persistent recurrent kernel: grid (2 dirs, 2 batch-halves) -------------
// Each WG owns 16 batches of one direction; batches never interact across
// time steps, so no cross-WG synchronization is needed.
__global__ __launch_bounds__(256)
void lstm_recurrent_kernel(const float* __restrict__ xg,
                           const __bf16* __restrict__ rkpack,
                           float* __restrict__ out) {
    // h (this WG's 16 batches) in A-fragment layout: [kt][lane][e] = 8 KB
    __shared__ __bf16 hA[KT * 32 * 16];
    // gate pre-activations: [col(1024)][local batch(16) + pad] = 80 KB
    __shared__ float gS[G4 * GPITCH];

    const int dir  = blockIdx.x;
    const int mh   = blockIdx.y;          // batch half: global rows mh*16..+15
    const int tid  = threadIdx.x;
    const int lane = tid & 31;
    const int w    = tid >> 5;

    for (int i = tid; i < KT * 32 * 16; i += 256) hA[i] = (__bf16)0.f;
    // c state: thread owns unit u = tid for its 16 local batches, in registers
    float c[16];
#pragma unroll
    for (int b = 0; b < 16; ++b) c[b] = 0.f;
    const int u      = tid;
    const int kt_u   = u >> 5;
    const int koff   = u & 31;
    const int laneHi = (koff >> 3) & 1;
    const int e_u    = (koff & 7) + ((koff >= 16) ? 8 : 0);
    __syncthreads();

    for (int s = 0; s < T_; ++s) {
        const int t_eff = dir ? (T_ - 1 - s) : s;

        // ---- Phase A: g = xg[t_eff] + h @ rk   (WMMA, 8 C-tiles per wave)
        for (int j = 0; j < 8; ++j) {
            int nt = w * 8 + j;
            v8f acc = *(const v8f*)(xg +
                ((((size_t)dir * T_ + t_eff) * MT + mh) * NT + nt) * 32 * 8 + lane * 8);
#pragma unroll
            for (int kt = 0; kt < KT; ++kt) {
                v16bf a = *(const v16bf*)(hA + (kt * 32 + lane) * 16);
                v16bf b = *(const v16bf*)(rkpack +
                    (((size_t)(dir * NT + nt) * KT + kt) * 32 + lane) * 16);
                acc = __builtin_amdgcn_wmma_f32_16x16x32_bf16(
                        false, a, false, b, (short)0, acc, false, false);
            }
            // C fragment rows are contiguous in [col][batch] layout: one v8f
            int col  = nt * 16 + (lane & 15);
            int mrow = 8 * (lane >> 4);
            *(v8f*)(gS + col * GPITCH + mrow) = acc;
        }
        // prefetch next step's xg fragments into cache while we sync
        if (s + 1 < T_) {
            int t_nx = dir ? (T_ - 2 - s) : (s + 1);
            __builtin_prefetch(xg +
                ((((size_t)dir * T_ + t_nx) * MT + mh) * NT) * 32 * 8 + tid * 8, 0, 0);
        }
        __syncthreads();

        // ---- Phase B: gates + state update; thread owns unit u, 16 batches
#pragma unroll
        for (int bb = 0; bb < 16; bb += 8) {
            v8f gi = *(const v8f*)(gS + (u         ) * GPITCH + bb);
            v8f gf = *(const v8f*)(gS + (U_     + u) * GPITCH + bb);
            v8f gm = *(const v8f*)(gS + (2 * U_ + u) * GPITCH + bb);
            v8f go = *(const v8f*)(gS + (3 * U_ + u) * GPITCH + bb);
#pragma unroll
            for (int r = 0; r < 8; ++r) {
                int b = bb + r;
                float i_ = hard_sig(gi[r]);
                float f_ = hard_sig(gf[r]);
                float m_ = fast_tanh(gm[r]);
                float o_ = hard_sig(go[r]);
                float cn = f_ * c[b] + i_ * m_;
                c[b] = cn;
                float hn = o_ * fast_tanh(cn);
                int bg = mh * 16 + b;
                out[((size_t)bg * T_ + t_eff) * (2 * U_) + dir * U_ + u] = hn;
                // write h into A-fragment LDS for next step's WMMA
                int ln = b + (laneHi << 4);
                hA[(kt_u * 32 + ln) * 16 + e_u] = (__bf16)hn;
            }
        }
        __syncthreads();
    }
}

extern "C" void kernel_launch(void* const* d_in, const int* in_sizes, int n_in,
                              void* d_out, int out_size, void* d_ws, size_t ws_size,
                              hipStream_t stream) {
    const float* x  = (const float*)d_in[0];   // [32,1024,256]
    const float* k  = (const float*)d_in[1];   // [256,2048]
    const float* rk = (const float*)d_in[2];   // [256,2048]
    float* out = (float*)d_out;                // [32,1024,512]

    char* ws = (char*)d_ws;
    // workspace layout:
    //   [0,16MB)       xA pack (bf16, 8,388,608 elems)
    //   [16,17MB)      kpack   (bf16, 524,288 elems)
    //   [17,18MB)      rkpack  (bf16, 524,288 elems)
    //   [18MB,+256MB)  xg      (f32, 67,108,864 elems, C-fragment order)
    __bf16* xA     = (__bf16*)(ws);
    __bf16* kpack  = (__bf16*)(ws + (size_t)16 * 1024 * 1024);
    __bf16* rkpack = (__bf16*)(ws + (size_t)17 * 1024 * 1024);
    float*  xg     = (float*) (ws + (size_t)18 * 1024 * 1024);

    hipLaunchKernelGGL(pack_w_kernel, dim3(256), dim3(256), 0, stream,
                       k, rk, kpack, rkpack);
    hipLaunchKernelGGL(pack_x_kernel, dim3(2048), dim3(256), 0, stream, x, xA);
    hipLaunchKernelGGL(xg_gemm_kernel, dim3(T_, 2), dim3(256), 0, stream,
                       xA, kpack, xg);
    hipLaunchKernelGGL(lstm_recurrent_kernel, dim3(2, 2), dim3(256), 0, stream,
                       xg, rkpack, out);
}
